// _Attention_128849018943
// MI455X (gfx1250) — matrix operand order repaired and checked by reading, not compiled, not run on hardware
//
#include <hip/hip_runtime.h>

// ---------------------------------------------------------------------------
// CDNA5 (gfx1250) causal MHA: QKV proj + flash attention + out proj.
// Matmuls: v_wmma_f32_16x16x32_bf16 (wave32).  Operand tiles without a
// transpose requirement are staged LDS-side by the Tensor Data Mover
// (tensor_load_to_lds, TENSORcnt), transpose tiles go global->VGPR->ds_store.
// ---------------------------------------------------------------------------

typedef __attribute__((ext_vector_type(16))) __bf16        v16bf;
typedef __attribute__((ext_vector_type(8)))  float         v8f;
typedef __attribute__((ext_vector_type(4)))  unsigned int  u32x4;
typedef __attribute__((ext_vector_type(8)))  unsigned int  u32x8;

union Frag16 { v16bf v; unsigned int u[8]; };   // A/B fragment, 8 VGPRs
union FragC  { v8f   v; float        f[8]; };   // C/D fragment, 8 VGPRs

#define SEQ    2048
#define DMODEL 1024
#define NHEAD  16
#define DHEAD  64

__device__ __forceinline__ unsigned short f2bf(float f) {
    unsigned int u = __builtin_bit_cast(unsigned int, f);
    u = (u + 0x7FFFu + ((u >> 16) & 1u)) >> 16;     // round-to-nearest-even
    return (unsigned short)u;
}

// ---------------------------------------------------------------------------
// TDM: DMA a 2D bf16 tile (tile_h rows x tile_w elems, row stride in elems)
// from global memory into LDS (packed row-major, no padding).
// D# layout per CDNA5 ISA ch.8 (group0 128b + group1 256b, 2D form).
// Issue from ONE wave only; complete with s_wait_tensorcnt 0 + barrier.
// ---------------------------------------------------------------------------
__device__ __forceinline__ void tdm_load_tile_2d(unsigned lds_off,
                                                 const void* gsrc,
                                                 unsigned tile_w_elems,
                                                 unsigned tile_h_rows,
                                                 unsigned row_stride_elems)
{
    unsigned long long ga = (unsigned long long)(uintptr_t)gsrc;
    u32x4 g0;
    g0[0] = 1u;                                   // count=1, user descriptor
    g0[1] = lds_off;                              // lds_addr (bytes)
    g0[2] = (unsigned)ga;                         // global_addr[31:0]
    g0[3] = (unsigned)((ga >> 32) & 0x01FFFFFFu)  // global_addr[56:32]
          | (2u << 30);                           // type = 2 ("image")
    const unsigned td0 = 1u << 20;                // generous tensor dims:
    const unsigned td1 = 1u << 20;                // tiles are always in-bounds
    u32x8 g1;
    g1[0] = 1u << 16;                             // data_size = 1 -> 2 bytes
    g1[1] = (td0 & 0xFFFFu) << 16;                // tensor_dim0[15:0]
    g1[2] = (td0 >> 16) | ((td1 & 0xFFFFu) << 16);
    g1[3] = (td1 >> 16) | ((tile_w_elems & 0xFFFFu) << 16);   // tile_dim0
    g1[4] = tile_h_rows & 0xFFFFu;                // tile_dim1 (tile_dim2 = 0)
    g1[5] = row_stride_elems;                     // tensor_dim0_stride[31:0]
    g1[6] = 0u;
    g1[7] = 0u;
    asm volatile("tensor_load_to_lds %0, %1" :: "s"(g0), "s"(g1) : "memory");
}

__device__ __forceinline__ unsigned lds_offset_of(const void* shared_ptr) {
    // generic LDS aperture address: addr[31:0] is the LDS byte offset
    return (unsigned)(uintptr_t)shared_ptr;
}

// ---------------------------------------------------------------------------
// fp32 -> bf16 convert
// ---------------------------------------------------------------------------
__global__ __launch_bounds__(256) void cvt_f32_bf16(const float* __restrict__ src,
                                                    unsigned short* __restrict__ dst,
                                                    int n) {
    int i = blockIdx.x * 256 + threadIdx.x;
    if (i < n) dst[i] = f2bf(src[i]);
}

// ---------------------------------------------------------------------------
// bf16 GEMM: C[M,N] = A[M,K] @ B[K,N] (both row-major), f32 accum.
// Block 128x64, 256 threads (8 waves = 4(M) x 2(N)), wave tile 32x32, K step 32.
// A tile (128x32) staged by TDM; B tile staged transposed manually (its
// fragment needs K-pairs contiguous).  STORE_BF16 templated (clean epilogue).
// ---------------------------------------------------------------------------
template <int STORE_BF16>
__global__ __launch_bounds__(256) void gemm_bf16_kernel(
    const unsigned short* __restrict__ A, const unsigned short* __restrict__ B,
    void* __restrict__ Cout, int M, int N, int K, float alphaScale)
{
    __shared__ unsigned short smem[128 * 32 + 64 * 36];
    unsigned short* At     = smem;                 // [128][32] row-major (TDM dest)
    unsigned short* BtBase = smem + 128 * 32;      // [64][36] transposed B tile

    const int tid  = threadIdx.x;
    const int wave = tid >> 5;
    const int lane = tid & 31;
    const int hlf  = lane >> 4;
    const int ln   = lane & 15;
    const int wm   = wave & 3;      // wave row (M)
    const int wn   = wave >> 2;     // wave col (N)

    const int bm0 = blockIdx.y * 128;
    const int bn0 = blockIdx.x * 64;
    const unsigned atOff = lds_offset_of(At);

    FragC acc[2][2];
    for (int mt = 0; mt < 2; ++mt)
        for (int nt = 0; nt < 2; ++nt)
            for (int i = 0; i < 8; ++i) acc[mt][nt].f[i] = 0.0f;

    for (int kc = 0; kc < K; kc += 32) {
        __syncthreads();
        if (wave == 0) {
            // DMA A[bm0..bm0+127][kc..kc+31] -> At (packed 128x32)
            tdm_load_tile_2d(atOff, &A[(size_t)bm0 * K + kc], 32, 128, (unsigned)K);
        }
        {   // stage B[kc..kc+31][bn0..bn0+63] transposed: 256 thr x 8 elems
            int k = tid >> 3;
            int n = (tid & 7) * 8;
            const uint4 d = *reinterpret_cast<const uint4*>(
                &B[(size_t)(kc + k) * N + bn0 + n]);
            unsigned short* bt = BtBase + k;       // column k of Bt
            bt[(n + 0) * 36] = (unsigned short)(d.x);
            bt[(n + 1) * 36] = (unsigned short)(d.x >> 16);
            bt[(n + 2) * 36] = (unsigned short)(d.y);
            bt[(n + 3) * 36] = (unsigned short)(d.y >> 16);
            bt[(n + 4) * 36] = (unsigned short)(d.z);
            bt[(n + 5) * 36] = (unsigned short)(d.z >> 16);
            bt[(n + 6) * 36] = (unsigned short)(d.w);
            bt[(n + 7) * 36] = (unsigned short)(d.w >> 16);
            if (kc + 32 < K)
                __builtin_prefetch(&B[(size_t)(kc + 32 + k) * N + bn0 + n], 0, 1);
        }
        if (wave == 0) __builtin_amdgcn_s_wait_tensorcnt(0);
        __syncthreads();

        Frag16 a[2];
        for (int mt = 0; mt < 2; ++mt)
            for (int i = 0; i < 8; ++i) {
                // A 16x32 layout: VGPR i holds K pair (i/4)*16 + half*8 + (i%4)*2
                int k = (i >> 2) * 16 + hlf * 8 + (i & 3) * 2;
                int r = wm * 32 + mt * 16 + ln;    // block-local row
                a[mt].u[i] = *reinterpret_cast<const unsigned int*>(&At[r * 32 + k]);
            }
        Frag16 b[2];
        for (int nt = 0; nt < 2; ++nt)
            for (int i = 0; i < 8; ++i) {
                // B 32x16 layout: VGPR i holds K pair (i/4)*16 + half*8 + (i%4)*2
                int k = (i >> 2) * 16 + hlf * 8 + (i & 3) * 2;
                int c = wn * 32 + nt * 16 + ln;    // block-local col
                b[nt].u[i] = *reinterpret_cast<const unsigned int*>(
                    &BtBase[c * 36 + k]);
            }
        for (int mt = 0; mt < 2; ++mt)
            for (int nt = 0; nt < 2; ++nt)
                acc[mt][nt].v = __builtin_amdgcn_wmma_f32_16x16x32_bf16(
                    false, a[mt].v, false, b[nt].v,
                    (short)0, acc[mt][nt].v, false, false);
    }

    // C/D layout: lanes 0-15 -> rows i, lanes 16-31 -> rows 8+i; col = ln
    for (int mt = 0; mt < 2; ++mt)
        for (int nt = 0; nt < 2; ++nt) {
            int row0 = bm0 + wm * 32 + mt * 16 + hlf * 8;
            int col  = bn0 + wn * 32 + nt * 16 + ln;
            if (STORE_BF16) {
                unsigned short* p = (unsigned short*)Cout + (size_t)row0 * N + col;
                for (int i = 0; i < 8; ++i)
                    p[(size_t)i * N] = f2bf(acc[mt][nt].f[i] * alphaScale);
            } else {
                float* p = (float*)Cout + (size_t)row0 * N + col;
                for (int i = 0; i < 8; ++i)
                    p[(size_t)i * N] = acc[mt][nt].f[i] * alphaScale;
            }
        }
}

// ---------------------------------------------------------------------------
// Flash attention (causal), Q pre-scaled by 1/sqrt(DH).
// Grid (S/64, H), 128 threads = 4 waves, each wave owns 16 query rows.
// KV chunk = 32: K tile (32x64) staged by TDM (K-dim contiguous -> B-frag reads
// straight from LDS); V tile staged transposed for the P@V WMMA; P bounced
// through per-wave LDS to convert C-layout -> A-layout.
// ---------------------------------------------------------------------------
__global__ __launch_bounds__(128) void attn_kernel(
    const unsigned short* __restrict__ Q, const unsigned short* __restrict__ Kmat,
    const unsigned short* __restrict__ V, unsigned short* __restrict__ Ctx)
{
    __shared__ unsigned short smem[2048 /*Vt*/ + 2048 /*Pl*/ + 2048 /*Kt*/];
    unsigned short* Vt = smem;               // [dh][kv]   stride 32
    unsigned short* Pl = smem + 2048;        // [wave][m][kv] stride 32
    unsigned short* Kt = smem + 4096;        // [kv][dh]   stride 64 (TDM dest)

    const int tid  = threadIdx.x;
    const int wave = tid >> 5;
    const int lane = tid & 31;
    const int hlf  = lane >> 4;
    const int ln   = lane & 15;

    const int head  = blockIdx.y;
    const int hOff  = head * DHEAD;
    const int bRow0 = blockIdx.x * 64;
    const int qRow0 = bRow0 + wave * 16;
    const unsigned ktOff = lds_offset_of(Kt);

    // Q fragments: 16 x 64 = two 16x32 A-frags (persistent in VGPRs)
    Frag16 qf[2];
    for (int kc2 = 0; kc2 < 2; ++kc2)
        for (int i = 0; i < 8; ++i) {
            int k = kc2 * 32 + ((i >> 2) * 16 + hlf * 8 + (i & 3) * 2);
            qf[kc2].u[i] = *reinterpret_cast<const unsigned int*>(
                &Q[(size_t)(qRow0 + ln) * DMODEL + hOff + k]);
        }

    float mrow[8], lrow[8];
    FragC o[4];
    for (int i = 0; i < 8; ++i) { mrow[i] = -1e30f; lrow[i] = 0.0f; }
    for (int nt = 0; nt < 4; ++nt)
        for (int i = 0; i < 8; ++i) o[nt].f[i] = 0.0f;

    const int kvEnd = bRow0 + 63;
    for (int kv0 = 0; kv0 <= kvEnd; kv0 += 32) {
        __syncthreads();
        if (wave == 0) {
            // DMA K[kv0..kv0+31][hOff..hOff+63] -> Kt (packed 32x64)
            tdm_load_tile_2d(ktOff, &Kmat[(size_t)kv0 * DMODEL + hOff],
                             64, 32, DMODEL);
        }
        // stage V^T chunk: 32 kv rows x 64 dh, 128 thr x 16 elems
        for (int j = 0; j < 2; ++j) {
            int L  = (tid * 2 + j) * 8;
            int kv = L >> 6;
            int dh = L & 63;
            const uint4 d = *reinterpret_cast<const uint4*>(
                &V[(size_t)(kv0 + kv) * DMODEL + hOff + dh]);
            unsigned short* vt = Vt + kv;        // column kv of Vt
            vt[(dh + 0) * 32] = (unsigned short)(d.x);
            vt[(dh + 1) * 32] = (unsigned short)(d.x >> 16);
            vt[(dh + 2) * 32] = (unsigned short)(d.y);
            vt[(dh + 3) * 32] = (unsigned short)(d.y >> 16);
            vt[(dh + 4) * 32] = (unsigned short)(d.z);
            vt[(dh + 5) * 32] = (unsigned short)(d.z >> 16);
            vt[(dh + 6) * 32] = (unsigned short)(d.w);
            vt[(dh + 7) * 32] = (unsigned short)(d.w >> 16);
            if (kv0 + 32 <= kvEnd)
                __builtin_prefetch(&V[(size_t)(kv0 + 32 + kv) * DMODEL + hOff + dh], 0, 1);
        }
        if (wave == 0) __builtin_amdgcn_s_wait_tensorcnt(0);
        __syncthreads();

        if (kv0 <= qRow0 + 15) {   // wave-uniform: EXEC stays all-ones inside
            // ---- scores: two 16x16 D tiles over the kv chunk ----
            FragC s[2];
            for (int t = 0; t < 2; ++t)
                for (int i = 0; i < 8; ++i) s[t].f[i] = 0.0f;
            for (int kvt = 0; kvt < 2; ++kvt)
                for (int kc2 = 0; kc2 < 2; ++kc2) {
                    Frag16 bk;   // K^T frag from Kt: dh pairs contiguous
                    for (int i = 0; i < 8; ++i) {
                        int kvloc = kvt * 16 + ln;
                        int k = kc2 * 32 + (i >> 2) * 16 + hlf * 8 + (i & 3) * 2;
                        bk.u[i] = *reinterpret_cast<const unsigned int*>(
                            &Kt[kvloc * 64 + k]);
                    }
                    s[kvt].v = __builtin_amdgcn_wmma_f32_16x16x32_bf16(
                        false, qf[kc2].v, false, bk.v,
                        (short)0, s[kvt].v, false, false);
                }

            // ---- causal mask + online softmax ----
            float alpha[8];
            for (int i = 0; i < 8; ++i) {
                int rowm = hlf * 8 + i;
                int qi = qRow0 + rowm;
                float mx = -1e30f;
                for (int kvt = 0; kvt < 2; ++kvt) {
                    int kv = kv0 + kvt * 16 + ln;
                    float sv = (kv <= qi) ? s[kvt].f[i] : -1e30f;
                    s[kvt].f[i] = sv;
                    mx = fmaxf(mx, sv);
                }
                for (int off = 1; off < 16; off <<= 1)
                    mx = fmaxf(mx, __shfl_xor(mx, off, 32));
                float mnew = fmaxf(mrow[i], mx);
                alpha[i] = __expf(mrow[i] - mnew);
                float ps = 0.0f;
                for (int kvt = 0; kvt < 2; ++kvt) {
                    float p = __expf(s[kvt].f[i] - mnew);
                    s[kvt].f[i] = p;
                    ps += p;
                }
                for (int off = 1; off < 16; off <<= 1)
                    ps += __shfl_xor(ps, off, 32);
                lrow[i] = lrow[i] * alpha[i] + ps;
                mrow[i] = mnew;
            }
            for (int nt = 0; nt < 4; ++nt)
                for (int i = 0; i < 8; ++i) o[nt].f[i] *= alpha[i];

            // ---- P: C-layout -> A-layout via per-wave LDS bounce ----
            unsigned short* pw = Pl + wave * 512;
            for (int kvt = 0; kvt < 2; ++kvt)
                for (int i = 0; i < 8; ++i)
                    pw[(hlf * 8 + i) * 32 + kvt * 16 + ln] = f2bf(s[kvt].f[i]);
            Frag16 pa;
            for (int i = 0; i < 8; ++i) {
                int k = (i >> 2) * 16 + hlf * 8 + (i & 3) * 2;
                pa.u[i] = *reinterpret_cast<const unsigned int*>(&pw[ln * 32 + k]);
            }

            // ---- O += P @ V ----
            for (int nt = 0; nt < 4; ++nt) {
                Frag16 bv;
                for (int i = 0; i < 8; ++i)
                    bv.u[i] = *reinterpret_cast<const unsigned int*>(
                        &Vt[(nt * 16 + ln) * 32 + (i >> 2) * 16 + hlf * 8 + (i & 3) * 2]);
                o[nt].v = __builtin_amdgcn_wmma_f32_16x16x32_bf16(
                    false, pa.v, false, bv.v, (short)0, o[nt].v, false, false);
            }
        }
    }

    // normalize and store context (bf16, [S, H*DH])
    for (int nt = 0; nt < 4; ++nt)
        for (int i = 0; i < 8; ++i) {
            int rowm = hlf * 8 + i;
            float vv = o[nt].f[i] / lrow[i];
            Ctx[(size_t)(qRow0 + rowm) * DMODEL + hOff + nt * 16 + ln] = f2bf(vv);
        }
}

// ---------------------------------------------------------------------------
// Host-side launch
// ---------------------------------------------------------------------------
extern "C" void kernel_launch(void* const* d_in, const int* in_sizes, int n_in,
                              void* d_out, int out_size, void* d_ws, size_t ws_size,
                              hipStream_t stream) {
    (void)in_sizes; (void)n_in; (void)out_size; (void)ws_size;
    const float* x  = (const float*)d_in[0];
    // d_in[1] = mask (causal, known analytically -> unused)
    const float* wq = (const float*)d_in[2];
    const float* wk = (const float*)d_in[3];
    const float* wv = (const float*)d_in[4];
    const float* wo = (const float*)d_in[5];

    const size_t nX = (size_t)SEQ * DMODEL;       // 2M elems
    const size_t nW = (size_t)DMODEL * DMODEL;    // 1M elems

    unsigned short* ws  = (unsigned short*)d_ws;
    unsigned short* xb  = ws;
    unsigned short* wqb = xb  + nX;
    unsigned short* wkb = wqb + nW;
    unsigned short* wvb = wkb + nW;
    unsigned short* wob = wvb + nW;
    unsigned short* Qb  = wob + nW;
    unsigned short* Kb  = Qb  + nX;
    unsigned short* Vb  = Kb  + nX;
    unsigned short* Cx  = Vb  + nX;

    cvt_f32_bf16<<<dim3((unsigned)((nX + 255) / 256)), 256, 0, stream>>>(x,  xb,  (int)nX);
    cvt_f32_bf16<<<dim3((unsigned)((nW + 255) / 256)), 256, 0, stream>>>(wq, wqb, (int)nW);
    cvt_f32_bf16<<<dim3((unsigned)((nW + 255) / 256)), 256, 0, stream>>>(wk, wkb, (int)nW);
    cvt_f32_bf16<<<dim3((unsigned)((nW + 255) / 256)), 256, 0, stream>>>(wv, wvb, (int)nW);
    cvt_f32_bf16<<<dim3((unsigned)((nW + 255) / 256)), 256, 0, stream>>>(wo, wob, (int)nW);

    dim3 gGemm(DMODEL / 64, SEQ / 128);
    gemm_bf16_kernel<1><<<gGemm, 256, 0, stream>>>(xb, wqb, (void*)Qb,
                                                   SEQ, DMODEL, DMODEL, 0.125f);
    gemm_bf16_kernel<1><<<gGemm, 256, 0, stream>>>(xb, wkb, (void*)Kb,
                                                   SEQ, DMODEL, DMODEL, 1.0f);
    gemm_bf16_kernel<1><<<gGemm, 256, 0, stream>>>(xb, wvb, (void*)Vb,
                                                   SEQ, DMODEL, DMODEL, 1.0f);

    attn_kernel<<<dim3(SEQ / 64, NHEAD), 128, 0, stream>>>(Qb, Kb, Vb, Cx);

    gemm_bf16_kernel<0><<<gGemm, 256, 0, stream>>>(Cx, wob, (void*)d_out,
                                                   SEQ, DMODEL, DMODEL, 1.0f);
}
